// CustomLSTM_12635793785203
// MI455X (gfx1250) — matrix operand order repaired and checked by reading, not compiled, not run on hardware
//
#include <hip/hip_runtime.h>

// ---------------------------------------------------------------------------
// Persistent WMMA LSTM for MI455X (gfx1250, wave32).
//   B=64, T=1024, I=256, H=512.
//   32 persistent workgroups x 128 threads (4 waves). WG g owns hidden
//   columns [g*16, g*16+16). Wave w handles M-tile w (rows w*16..w*16+15) and
//   all four 16-wide gate tiles (i,f,g,o) for the WG's hidden slice.
//   Weights live in LDS (f16, column-major, combined K = [U(256);V(512)]).
//   x_t tiles double-buffered in LDS via GLOBAL_LOAD_ASYNC_TO_LDS_B128.
//   Combined-K GEMM loop is software-pipelined: B frags 1 iter ahead (LDS),
//   A frags 2 iters ahead (LDS x-tile, then global h). First iteration is
//   peeled so WMMA consumes bias splats as C (no per-step acc-init movs).
//   h is double-buffered f16 in global workspace (L2-resident); one
//   device-wide barrier per timestep.
// ---------------------------------------------------------------------------

typedef __attribute__((ext_vector_type(16))) _Float16 v16h;
typedef __attribute__((ext_vector_type(8)))  _Float16 v8h;
typedef __attribute__((ext_vector_type(8)))  float    v8f;

#define BB   64
#define TT   1024
#define II   256
#define HH   512
#define NWG  32
#define HS   16            // hidden columns per workgroup
#define NTHR 128           // 4 waves
#define KTOT (II + HH)     // 768 combined K
#define NK   (KTOT / 32)   // 24 WMMA k-steps
#define WSTRIDE 776        // weight panel column stride (768 + 8 pad)
#define WPANEL_HALFS (64 * WSTRIDE)            // 49,664 halfs
#define XROW 264           // x tile row stride in halfs (256 + 8 pad)
#define XBUF_HALFS (BB * XROW)                 // 16,896 halfs per buffer
#define LDS_HALFS (WPANEL_HALFS + 2 * XBUF_HALFS)
#define LDS_BYTES (LDS_HALFS * 2)              // 166,912 B < 320 KB
#define X16_HALFS  (TT * BB * II)              // 32 MB staging of x in f16
#define HBUF_HALFS (2 * BB * HH)               // 128 KB double-buffered h

#define CAT16(LO, HI) \
  __builtin_shufflevector(LO, HI, 0,1,2,3,4,5,6,7,8,9,10,11,12,13,14,15)

__device__ __forceinline__ float fast_sigmoid(float v) {
  return 1.0f / (1.0f + __expf(-v));
}
__device__ __forceinline__ float fast_tanh(float v) {
  return 2.0f / (1.0f + __expf(-2.0f * v)) - 1.0f;   // exact identity
}
__device__ __forceinline__ v8f wmma_f16(v16h a, v16h b, v8f c) {
  return __builtin_amdgcn_wmma_f32_16x16x32_f16(false, a, false, b,
                                                (short)0, c, false, false);
}

// Device-wide barrier: monotonic counter, agent-scope fences. Counter is
// reset by lstm_init every launch, so graph replays are deterministic.
__device__ __forceinline__ void grid_barrier(unsigned* cnt) {
  __syncthreads();
  if (threadIdx.x == 0) {
    __builtin_amdgcn_fence(__ATOMIC_RELEASE, "agent");
    unsigned a = __hip_atomic_fetch_add(cnt, 1u, __ATOMIC_RELAXED,
                                        __HIP_MEMORY_SCOPE_AGENT);
    unsigned target = (a / NWG + 1u) * NWG;
    while (__hip_atomic_load(cnt, __ATOMIC_RELAXED,
                             __HIP_MEMORY_SCOPE_AGENT) < target)
      __builtin_amdgcn_s_sleep(2);
  }
  __syncthreads();
  __builtin_amdgcn_fence(__ATOMIC_ACQUIRE, "agent");  // invalidate stale h
}

__global__ void lstm_init(_Float16* hbuf, unsigned* bar) {
  int tid = blockIdx.x * blockDim.x + threadIdx.x;
  if (tid == 0) *bar = 0u;
  for (int i = tid; i < HBUF_HALFS; i += gridDim.x * blockDim.x)
    hbuf[i] = (_Float16)0.0f;
}

__global__ void __launch_bounds__(NTHR, 1)
lstm_persistent(const float* __restrict__ x,
                const float* __restrict__ Ui, const float* __restrict__ Vi, const float* __restrict__ bi,
                const float* __restrict__ Uf, const float* __restrict__ Vf, const float* __restrict__ bf,
                const float* __restrict__ Uh, const float* __restrict__ Vh, const float* __restrict__ bh,
                const float* __restrict__ Uo, const float* __restrict__ Vo, const float* __restrict__ bo,
                const float* __restrict__ fcw, const float* __restrict__ fcb,
                _Float16* __restrict__ x16,
                _Float16* __restrict__ hbuf,
                unsigned* __restrict__ bar,
                float* __restrict__ out)
{
  extern __shared__ _Float16 lds[];
  const int tid  = threadIdx.x;
  const int g    = blockIdx.x;     // hidden-slice owner
  const int lane = tid & 31;
  const int m    = tid >> 5;       // wave id == M-tile id (0..3)
  const int nlo  = lane & 15;
  const int hi16 = lane >> 4;

  // ---- one-time: stage this WG's weight panel into LDS (f32 -> f16) ----
  {
    const float* Us[4] = {Ui, Uf, Uh, Uo};
    const float* Vs[4] = {Vi, Vf, Vh, Vo};
    for (int idx = tid; idx < 64 * KTOT; idx += NTHR) {
      int j    = idx / KTOT;             // column 0..63 = gate*16 + jc
      int r    = idx - j * KTOT;         // combined K row 0..767
      int gate = j >> 4;
      int col  = g * HS + (j & 15);      // column within H
      float v = (r < II) ? Us[gate][r * HH + col]
                         : Vs[gate][(r - II) * HH + col];
      lds[j * WSTRIDE + r] = (_Float16)v;
    }
  }

  // ---- one-time: pack x fp32 [B,T,I] -> f16 [T,B,I]; no divides, coalesced
  {
    const int gid = g * NTHR + tid;                    // 0..4095
    for (int t = 0; t < TT; ++t) {
      #pragma unroll
      for (int q = gid; q < BB * II; q += NWG * NTHR) { // 4 iterations
        int b = q >> 8;
        int i = q & 255;
        x16[t * (BB * II) + q] = (_Float16)x[(b * TT + t) * II + i];
      }
    }
  }

  grid_barrier(bar);

  const int hc = g * HS + nlo;                       // this lane's hidden col
  const int arow = m * 16 + nlo;                     // A-fragment row (M)
  const int akg  = hi16 * 8;                         // A: split-8 K groups
  const int bkg  = hi16 * 8;                         // B: split-8 K groups

  // Bias splats used as the C operand of the first (peeled) WMMA per step.
  v8f biasv[4];
  {
    const float b0 = bi[hc], b1 = bf[hc], b2 = bh[hc], b3 = bo[hc];
    #pragma unroll
    for (int r = 0; r < 8; ++r) {
      biasv[0][r] = b0; biasv[1][r] = b1; biasv[2][r] = b2; biasv[3][r] = b3;
    }
  }

  // Async-copy one 64x256 f16 x tile (this wave's 16 rows) into LDS buffer.
  // 16 issues per lane per tile -> per-wave ASYNCcnt pipelining.
  auto issue_x_tile = [&](int t, int buf) {
    _Float16* dst = lds + WPANEL_HALFS + buf * XBUF_HALFS;
    const _Float16* gbase = x16 + (t * BB + m * 16) * II;
    #pragma unroll
    for (int q = lane; q < 512; q += 32) {           // 16 iterations
      int row   = q >> 5;                            // 0..15
      int chunk = q & 31;                            // 32 x 16B per row
      unsigned loff =
          (unsigned)(unsigned long long)(dst + (m * 16 + row) * XROW + chunk * 8);
      const _Float16* ga = gbase + row * II + chunk * 8;
      asm volatile("global_load_async_to_lds_b128 %0, %1, off"
                   :: "v"(loff), "v"(ga) : "memory");
    }
  };

  issue_x_tile(0, 0);                                // prime the pipeline

  v8f c;
  #pragma unroll
  for (int r = 0; r < 8; ++r) c[r] = 0.0f;

  for (int t = 0; t < TT; ++t) {
    // Prefetch x_{t+1} into the other LDS buffer, then drain x_t's copies.
    if (t + 1 < TT) {
      issue_x_tile(t + 1, (t + 1) & 1);
      asm volatile("s_wait_asynccnt 0x10" ::: "memory");  // <=16: x_t done
    } else {
      asm volatile("s_wait_asynccnt 0x0" ::: "memory");
    }

    const _Float16* __restrict__ hread  = hbuf + (t & 1) * (BB * HH);
    _Float16*       __restrict__ hwrite = hbuf + ((t & 1) ^ 1) * (BB * HH);

    const _Float16* xtile = lds + WPANEL_HALFS + (t & 1) * XBUF_HALFS
                                + arow * XROW;
    const _Float16* __restrict__ hbase = hread + arow * HH;

    // A fragment over combined K: [0,256) from LDS x tile, [256,768) from h.
    auto loadA = [&](int k) -> v16h {
      const _Float16* p = (k < II) ? (xtile + k + akg)
                                   : (hbase + (k - II) + akg);
      return CAT16(*(const v8h*)p, *(const v8h*)(p + 16));
    };
    // B fragment: weight panel column (gate*16 + nlo), combined K row k.
    auto loadB = [&](int k, int gate) -> v16h {
      const _Float16* cb = lds + (gate * 16 + nlo) * WSTRIDE + k + bkg;
      return CAT16(*(const v8h*)cb, *(const v8h*)(cb + 16));
    };

    // Software-pipelined combined-K GEMM: B 1 ahead, A 2 ahead.
    v16h aC = loadA(0);
    v16h aN = loadA(32);
    v16h bC[4], bN[4];
    #pragma unroll
    for (int gq = 0; gq < 4; ++gq) bC[gq] = loadB(0, gq);

    v8f acc[4];
    #pragma unroll
    for (int ki = 0; ki < NK; ++ki) {
      const int k = ki * 32;
      if (ki + 1 < NK) {
        #pragma unroll
        for (int gq = 0; gq < 4; ++gq) bN[gq] = loadB(k + 32, gq);
      }
      v16h aNN = aN;
      if (ki + 2 < NK) aNN = loadA(k + 64);
      #pragma unroll
      for (int gq = 0; gq < 4; ++gq) {
        if (ki == 0)
          acc[gq] = wmma_f16(aC, bC[gq], biasv[gq]);   // D = A*B + bias
        else
          acc[gq] = wmma_f16(aC, bC[gq], acc[gq]);
      }
      aC = aN; aN = aNN;
      #pragma unroll
      for (int gq = 0; gq < 4; ++gq) bC[gq] = bN[gq];
    }

    // ---- gates + state update; D layout: VGPR r holds M = m*16 + r + 8*hi16
    #pragma unroll
    for (int r = 0; r < 8; ++r) {
      float iv = fast_sigmoid(acc[0][r]);
      float fv = fast_sigmoid(acc[1][r]);
      float gv = fast_tanh(acc[2][r]);
      float ov = fast_sigmoid(acc[3][r]);
      float cv = fv * c[r] + iv * gv;
      c[r] = cv;
      float hv = ov * fast_tanh(cv);
      int row = m * 16 + r + hi16 * 8;
      hwrite[row * HH + hc] = (_Float16)hv;
    }

    grid_barrier(bar);
  }

  // ---- final FC: out[b] = h_last[b,:] . fc_w + fc_b (TT even -> buffer 0)
  if (g == 0 && tid < BB) {
    const _Float16* hl = hbuf;
    float s = fcb[0];
    for (int k = 0; k < HH; ++k)
      s += (float)hl[tid * HH + k] * fcw[k];
    out[tid] = s;
  }
}

extern "C" void kernel_launch(void* const* d_in, const int* in_sizes, int n_in,
                              void* d_out, int out_size, void* d_ws, size_t ws_size,
                              hipStream_t stream) {
  const float* x   = (const float*)d_in[0];
  const float* Ui  = (const float*)d_in[1];
  const float* Vi  = (const float*)d_in[2];
  const float* bi  = (const float*)d_in[3];
  const float* Uf  = (const float*)d_in[4];
  const float* Vf  = (const float*)d_in[5];
  const float* bf  = (const float*)d_in[6];
  const float* Uh  = (const float*)d_in[7];
  const float* Vh  = (const float*)d_in[8];
  const float* bh  = (const float*)d_in[9];
  const float* Uo  = (const float*)d_in[10];
  const float* Vo  = (const float*)d_in[11];
  const float* bo  = (const float*)d_in[12];
  const float* fcw = (const float*)d_in[13];
  const float* fcb = (const float*)d_in[14];

  char* ws = (char*)d_ws;
  _Float16* x16  = (_Float16*)ws;
  _Float16* hbuf = (_Float16*)(ws + (size_t)X16_HALFS * 2);
  unsigned* bar  = (unsigned*)(ws + (size_t)X16_HALFS * 2 + (size_t)HBUF_HALFS * 2);

  (void)in_sizes; (void)n_in; (void)out_size; (void)ws_size;

  // Allow > default dynamic LDS (163 KB); idempotent, deterministic.
  hipFuncSetAttribute(reinterpret_cast<const void*>(lstm_persistent),
                      hipFuncAttributeMaxDynamicSharedMemorySize,
                      (int)LDS_BYTES);

  lstm_init<<<dim3(64), dim3(256), 0, stream>>>(hbuf, bar);
  lstm_persistent<<<dim3(NWG), dim3(NTHR), LDS_BYTES, stream>>>(
      x, Ui, Vi, bi, Uf, Vf, bf, Uh, Vh, bh, Uo, Vo, bo, fcw, fcb,
      x16, hbuf, bar, (float*)d_out);
}
